// Decoder_37022618091628
// MI455X (gfx1250) — matrix operand order repaired and checked by reading, not compiled, not run on hardware
//
#include <hip/hip_runtime.h>
#include <hip/hip_bf16.h>

// Problem constants (from reference): B=32, S=64, T=32, E=512, H=1024, V=32000
#define BB 32
#define SS 64
#define TT 32
#define EE 512
#define HH 1024
#define VV 32000
#define XDIM (EE + 2*HH + HH)   // 3584 : [emb | ctx | h] concat row

typedef unsigned short bfraw;   // raw bf16 storage for host-visible signatures
typedef __attribute__((ext_vector_type(16))) __bf16 v16bf;
typedef __attribute__((ext_vector_type(8)))  __bf16 v8bf;
typedef __attribute__((ext_vector_type(8)))  float  v8f;

// ----------------------------------------------------------------------------
// fp32 -> bf16 elementwise conversion
// ----------------------------------------------------------------------------
__global__ void k_f32_to_bf16(const float* __restrict__ src, bfraw* __restrict__ dst, int n) {
    int i = blockIdx.x * 256 + threadIdx.x;
    if (i < n) reinterpret_cast<__bf16*>(dst)[i] = (__bf16)src[i];
}

// ----------------------------------------------------------------------------
// Pack fp32 weight matrix into the per-lane WMMA B-operand layout (bf16).
// Packed index: (((nt*KT_dest + ktOff+kt)*32 + lane)*16 + e)
// where n = nt*16 + (lane&15), k = kt*32 + (lane>=16 ? 16 : 0) + e
// mode 0: src is [K,N] row-major             (val = src[k*N+n])
// mode 1: src is [N,Ksrc] row-major (transp) (val = src[n*srcStride+k])
// mode 2: like mode 0 but with W_pre row permutation mapping [emb|h|ctx]
//         source order onto our [emb|ctx|h] activation buffer order.
// ----------------------------------------------------------------------------
__global__ void k_pack_b(const float* __restrict__ src, bfraw* __restrict__ dstR,
                         int N, int KT_dest, int ktOff, int KTregion,
                         int srcStride, int mode) {
    __bf16* dst = reinterpret_cast<__bf16*>(dstR);
    int gid  = blockIdx.x * 256 + threadIdx.x;
    int tile = gid >> 5, lane = gid & 31;
    int tiles = KTregion * (N >> 4);
    if (tile >= tiles) return;
    int kt = tile % KTregion, nt = tile / KTregion;
    int n  = (nt << 4) + (lane & 15);
    int kb = (kt << 5) + ((lane >> 4) << 4);
    __bf16* d = dst + ((((size_t)nt * KT_dest + (ktOff + kt)) * 32) + lane) * 16;
#pragma unroll
    for (int e = 0; e < 16; ++e) {
        int k = kb + e;
        float v;
        if (mode == 1) {
            v = src[(size_t)n * srcStride + k];
        } else if (mode == 2) {
            int ks = (k < EE) ? k : ((k < EE + 2*HH) ? (k + HH) : (k - 2*HH));
            v = src[(size_t)ks * N + n];
        } else {
            v = src[(size_t)k * N + n];
        }
        d[e] = (__bf16)v;
    }
}

// ----------------------------------------------------------------------------
// Generic WMMA GEMM: C[M,N] = A[M,K](bf16,row-major,lda) * Bpacked[K,N](bf16)
// Block = 256 threads = 8 waves; each wave computes a 32x16 strip (two 16x16
// m-tiles sharing one B fragment -> B stream traffic halved, 3 loads / wmma).
// K-loop is 2-stage software pipelined: fragments for k+1 are loaded before
// the wmmas for k so loads stay outstanding across the matrix ops (partial
// s_wait_loadcnt instead of full drains).
// Block tile 32x128: grid.x = N/128, grid.y = M/32. K multiple of 32.
// EPI: 0 = f32 out, 1 = f32 out + bias[col], 2 = bf16 out
// ----------------------------------------------------------------------------
template <int EPI>
__global__ __launch_bounds__(256)
void k_gemm(const bfraw* __restrict__ Ar, int lda,
            const bfraw* __restrict__ Bpr, int KT, int N,
            void* __restrict__ Cout, int ldc,
            const float* __restrict__ bias) {
    const __bf16* A  = reinterpret_cast<const __bf16*>(Ar);
    const __bf16* Bp = reinterpret_cast<const __bf16*>(Bpr);
    const int lane  = threadIdx.x & 31;
    const int wave  = threadIdx.x >> 5;             // 0..7 -> n-tile within block
    const int mbase = blockIdx.y * 32;
    const int ntile = blockIdx.x * 8 + wave;        // global 16-column tile
    const int row   = mbase + (lane & 15);
    const int khA   = (lane >> 4) << 4;             // A half-K select per lane group

    const __bf16* a_ptr = A + (size_t)row * lda + khA;
    const __bf16* b_ptr = Bp + ((size_t)ntile * KT * 32) * 16 + (size_t)lane * 16;
    const size_t  aRow16 = (size_t)16 * lda;        // second m-tile offset

    v8f c0 = {}, c1 = {};

    // pipeline stage 0: preload kt=0 fragments
    v8bf b0  = *reinterpret_cast<const v8bf*>(b_ptr);
    v8bf b1  = *reinterpret_cast<const v8bf*>(b_ptr + 8);
    v8bf a00 = *reinterpret_cast<const v8bf*>(a_ptr);
    v8bf a01 = *reinterpret_cast<const v8bf*>(a_ptr + 8);
    v8bf a10 = *reinterpret_cast<const v8bf*>(a_ptr + aRow16);
    v8bf a11 = *reinterpret_cast<const v8bf*>(a_ptr + aRow16 + 8);

    for (int kt = 0; kt < KT - 1; ++kt) {
        a_ptr += 32;
        b_ptr += 512;                               // 32 lanes * 16 elems
        // prefetch next-k fragments (kept outstanding across the wmmas below)
        v8bf nb0  = *reinterpret_cast<const v8bf*>(b_ptr);
        v8bf nb1  = *reinterpret_cast<const v8bf*>(b_ptr + 8);
        v8bf na00 = *reinterpret_cast<const v8bf*>(a_ptr);
        v8bf na01 = *reinterpret_cast<const v8bf*>(a_ptr + 8);
        v8bf na10 = *reinterpret_cast<const v8bf*>(a_ptr + aRow16);
        v8bf na11 = *reinterpret_cast<const v8bf*>(a_ptr + aRow16 + 8);

        v16bf av0, av1, bv;
#pragma unroll
        for (int i = 0; i < 8; ++i) {
            av0[i] = a00[i]; av0[i + 8] = a01[i];
            av1[i] = a10[i]; av1[i + 8] = a11[i];
            bv[i]  = b0[i];  bv[i + 8]  = b1[i];
        }
        c0 = __builtin_amdgcn_wmma_f32_16x16x32_bf16(false, av0, false, bv,
                                                     (short)0, c0, false, false);
        c1 = __builtin_amdgcn_wmma_f32_16x16x32_bf16(false, av1, false, bv,
                                                     (short)0, c1, false, false);
        b0 = nb0; b1 = nb1;
        a00 = na00; a01 = na01; a10 = na10; a11 = na11;
    }
    {   // drain: final k-tile
        v16bf av0, av1, bv;
#pragma unroll
        for (int i = 0; i < 8; ++i) {
            av0[i] = a00[i]; av0[i + 8] = a01[i];
            av1[i] = a10[i]; av1[i + 8] = a11[i];
            bv[i]  = b0[i];  bv[i + 8]  = b1[i];
        }
        c0 = __builtin_amdgcn_wmma_f32_16x16x32_bf16(false, av0, false, bv,
                                                     (short)0, c0, false, false);
        c1 = __builtin_amdgcn_wmma_f32_16x16x32_bf16(false, av1, false, bv,
                                                     (short)0, c1, false, false);
    }

    const int col = (ntile << 4) + (lane & 15);
    const int r0  = mbase + ((lane >> 4) << 3);     // C rows: lanes<16 -> M=e, else M=8+e
    float bv = (EPI == 1) ? bias[col] : 0.0f;
#pragma unroll
    for (int e = 0; e < 8; ++e) {
        float v0 = c0[e] + bv;
        float v1 = c1[e] + bv;
        if (EPI == 2) {
            reinterpret_cast<__bf16*>(Cout)[(size_t)(r0 + e) * ldc + col]      = (__bf16)v0;
            reinterpret_cast<__bf16*>(Cout)[(size_t)(r0 + 16 + e) * ldc + col] = (__bf16)v1;
        } else {
            reinterpret_cast<float*>(Cout)[(size_t)(r0 + e) * ldc + col]      = v0;
            reinterpret_cast<float*>(Cout)[(size_t)(r0 + 16 + e) * ldc + col] = v1;
        }
    }
}

// ----------------------------------------------------------------------------
// Attention energies: scores[b,s] = sum_h tanh(q[b,h]+proj_key[b,s,h])*v[h],
// masked to -1e9 where s >= src_lengths[b]. One wave per (b,s).
// ----------------------------------------------------------------------------
__global__ void k_scores(const float* __restrict__ q, const float* __restrict__ pk,
                         const float* __restrict__ ve, const int* __restrict__ src_len,
                         float* __restrict__ scores) {
    int wid  = (blockIdx.x * 256 + threadIdx.x) >> 5;   // 0 .. B*S-1 (exact grid)
    int lane = threadIdx.x & 31;
    int b = wid >> 6, s = wid & 63;
    const float* qb  = q  + (size_t)b * HH;
    const float* pks = pk + (size_t)wid * HH;
    float acc = 0.f;
    for (int h = lane; h < HH; h += 32)
        acc += tanhf(qb[h] + pks[h]) * ve[h];
#pragma unroll
    for (int off = 16; off; off >>= 1) acc += __shfl_down(acc, off, 32);
    if (lane == 0) scores[wid] = (s < src_len[b]) ? acc : -1e9f;
}

// ----------------------------------------------------------------------------
// Softmax over S + context = alphas @ encoder_hidden; writes bf16 context
// directly into the ctx slot of the xh concat buffer. One block per batch row.
// ----------------------------------------------------------------------------
__global__ void k_softmax_ctx(const float* __restrict__ scores,
                              const float* __restrict__ eh,
                              bfraw* __restrict__ xhR) {
    __bf16* xh = reinterpret_cast<__bf16*>(xhR);
    __shared__ float sc[SS], al[SS];
    int b = blockIdx.x, tid = threadIdx.x;
    if (tid < SS) sc[tid] = scores[b * SS + tid];
    __syncthreads();
    float m = -1e30f;
    for (int s = 0; s < SS; ++s) m = fmaxf(m, sc[s]);
    float sum = 0.f;
    for (int s = 0; s < SS; ++s) sum += __expf(sc[s] - m);
    float inv = 1.f / sum;
    if (tid < SS) al[tid] = __expf(sc[tid] - m) * inv;
    __syncthreads();
    const float* ehb = eh + (size_t)b * SS * 2 * HH;
    for (int d = tid; d < 2 * HH; d += 256) {
        float acc = 0.f;
        for (int s = 0; s < SS; ++s) acc += al[s] * ehb[(size_t)s * 2 * HH + d];
        xh[(size_t)b * XDIM + EE + d] = (__bf16)acc;
    }
}

// ----------------------------------------------------------------------------
// Embedding gather -> bf16 emb slot of xh
// ----------------------------------------------------------------------------
__global__ void k_embed(const int* __restrict__ trg, const float* __restrict__ emb,
                        bfraw* __restrict__ xhR, int t) {
    __bf16* xh = reinterpret_cast<__bf16*>(xhR);
    int b = blockIdx.x, tid = threadIdx.x;
    int y = trg[b * TT + t];
    for (int e = tid; e < EE; e += 256)
        xh[(size_t)b * XDIM + e] = (__bf16)emb[(size_t)y * EE + e];
}

// ----------------------------------------------------------------------------
// LSTM pointwise: gate order i,f,g,o; updates fp32 h,c and bf16 h slot of xh
// ----------------------------------------------------------------------------
__device__ __forceinline__ float sigm(float x) { return 1.f / (1.f + __expf(-x)); }

__global__ void k_lstm(const float* __restrict__ gates, float* __restrict__ h,
                       float* __restrict__ c, bfraw* __restrict__ xhR) {
    __bf16* xh = reinterpret_cast<__bf16*>(xhR);
    int idx = blockIdx.x * 256 + threadIdx.x;          // B*H
    int b = idx >> 10, j = idx & 1023;
    const float* g = gates + (size_t)b * 4 * HH;
    float gi = g[j], gf = g[HH + j], gg = g[2 * HH + j], go = g[3 * HH + j];
    float cn = sigm(gf) * c[idx] + sigm(gi) * tanhf(gg);
    float hn = sigm(go) * tanhf(cn);
    c[idx] = cn;
    h[idx] = hn;
    xh[(size_t)b * XDIM + EE + 2 * HH + j] = (__bf16)hn;
}

// ----------------------------------------------------------------------------
// Vocab softmax; out layout [B,V,T] (stride T per vocab entry). Block per b.
// ----------------------------------------------------------------------------
__global__ void k_softmax_v(const float* __restrict__ logits, float* __restrict__ out, int t) {
    __shared__ float red[256];
    int b = blockIdx.x, tid = threadIdx.x;
    const float* lb = logits + (size_t)b * VV;
    float m = -1e30f;
    for (int v = tid; v < VV; v += 256) m = fmaxf(m, lb[v]);
    red[tid] = m; __syncthreads();
    for (int s = 128; s; s >>= 1) { if (tid < s) red[tid] = fmaxf(red[tid], red[tid + s]); __syncthreads(); }
    m = red[0]; __syncthreads();
    float sum = 0.f;
    for (int v = tid; v < VV; v += 256) sum += __expf(lb[v] - m);
    red[tid] = sum; __syncthreads();
    for (int s = 128; s; s >>= 1) { if (tid < s) red[tid] += red[tid + s]; __syncthreads(); }
    float inv = 1.f / red[0];
    float* ob = out + (size_t)b * VV * TT + t;
    for (int v = tid; v < VV; v += 256) ob[(size_t)v * TT] = __expf(lb[v] - m) * inv;
}

// ----------------------------------------------------------------------------
// One-time bridge init: h0/c0 = tanh(ef @ W + b). Naive fp32 (134M MAC, once).
// xh_h (may be null) is the bf16 h-slot base (xh + EE + 2*HH offset applied by caller).
// ----------------------------------------------------------------------------
__global__ void k_init_state(const float* __restrict__ ef, const float* __restrict__ W,
                             const float* __restrict__ bias, float* __restrict__ dst,
                             bfraw* __restrict__ xh_hR) {
    int idx = blockIdx.x * 256 + threadIdx.x;          // B*H
    int b = idx >> 10, j = idx & 1023;
    const float* x = ef + (size_t)b * 2 * HH;
    float acc = bias[j];
    for (int k = 0; k < 2 * HH; ++k) acc += x[k] * W[(size_t)k * HH + j];
    float v = tanhf(acc);
    dst[idx] = v;
    if (xh_hR) reinterpret_cast<__bf16*>(xh_hR)[(size_t)b * XDIM + j] = (__bf16)v;
}

__global__ void k_addvec(const float* __restrict__ a, const float* __restrict__ b,
                         float* __restrict__ o, int n) {
    int i = blockIdx.x * 256 + threadIdx.x;
    if (i < n) o[i] = a[i] + b[i];
}

__global__ void k_copy_hc(const float* __restrict__ h, const float* __restrict__ c,
                          float* __restrict__ out) {
    int i = blockIdx.x * 256 + threadIdx.x;
    if (i < BB * HH) { out[i] = h[i]; out[BB * HH + i] = c[i]; }
}

// ----------------------------------------------------------------------------
extern "C" void kernel_launch(void* const* d_in, const int* in_sizes, int n_in,
                              void* d_out, int out_size, void* d_ws, size_t ws_size,
                              hipStream_t stream) {
    (void)in_sizes; (void)n_in; (void)out_size; (void)ws_size;
    const int*   trg     = (const int*)d_in[0];
    const int*   src_len = (const int*)d_in[1];
    const float* enc     = (const float*)d_in[2];
    const float* ef_h    = (const float*)d_in[3];
    const float* ef_c    = (const float*)d_in[4];
    const float* emb     = (const float*)d_in[5];
    const float* Wk      = (const float*)d_in[6];
    const float* Wq      = (const float*)d_in[7];
    const float* ve      = (const float*)d_in[8];
    const float* Wbh     = (const float*)d_in[9];
    const float* bbh     = (const float*)d_in[10];
    const float* Wbc     = (const float*)d_in[11];
    const float* bbc     = (const float*)d_in[12];
    const float* W_ih    = (const float*)d_in[13];
    const float* W_hh    = (const float*)d_in[14];
    const float* b_ih    = (const float*)d_in[15];
    const float* b_hh    = (const float*)d_in[16];
    const float* W_pre   = (const float*)d_in[17];
    const float* W_gen   = (const float*)d_in[18];
    const float* b_gen   = (const float*)d_in[19];
    float* out = (float*)d_out;

    // ---- workspace carve-out (deterministic) ----
    char*  ws  = (char*)d_ws;
    size_t off = 0;
    auto take = [&](size_t bytes) -> void* {
        void* p = ws + off;
        off = (off + bytes + 255) & ~(size_t)255;
        return p;
    };
    bfraw* pWq   = (bfraw*)take((size_t)HH * HH * 2);            // packed Wq   [1024,1024]
    bfraw* pWk   = (bfraw*)take((size_t)2 * HH * HH * 2);        // packed Wk   [2048,1024]
    bfraw* pWl   = (bfraw*)take((size_t)XDIM * 4 * HH * 2);      // packed LSTM [3584,4096]
    bfraw* pWpre = (bfraw*)take((size_t)XDIM * HH * 2);          // packed Wpre [3584,1024]
    bfraw* pWgen = (bfraw*)take((size_t)HH * VV * 2);            // packed Wgen [1024,32000]
    bfraw* ehB   = (bfraw*)take((size_t)BB * SS * 2 * HH * 2);   // bf16 encoder_hidden
    float* pkey  = (float*)take((size_t)BB * SS * HH * 4);       // proj_key fp32
    bfraw* xh    = (bfraw*)take((size_t)BB * XDIM * 2);          // [emb|ctx|h] bf16 activations
    float* hbuf  = (float*)take((size_t)BB * HH * 4);
    float* cbuf  = (float*)take((size_t)BB * HH * 4);
    float* qbuf  = (float*)take((size_t)BB * HH * 4);
    float* sbuf  = (float*)take((size_t)BB * SS * 4);
    float* gbuf  = (float*)take((size_t)BB * 4 * HH * 4);
    bfraw* preA  = (bfraw*)take((size_t)BB * HH * 2);
    float* lbuf  = (float*)take((size_t)BB * VV * 4);
    float* bsum  = (float*)take((size_t)4 * HH * 4);

    dim3 blk(256);
    auto pack = [&](const float* src, bfraw* dst, int N, int KTd, int ktOff,
                    int KTreg, int sStride, int mode) {
        int threads = KTreg * (N / 16) * 32;
        k_pack_b<<<dim3((threads + 255) / 256), blk, 0, stream>>>(
            src, dst, N, KTd, ktOff, KTreg, sStride, mode);
    };

    // ---- one-time setup: weight packing (bf16, WMMA-operand order; ~108MB, L2-resident) ----
    pack(Wq,    pWq,   HH,   32,  0,  32, 0,    0);
    pack(Wk,    pWk,   HH,   64,  0,  64, 0,    0);
    pack(W_ih,  pWl,   4*HH, 112, 0,  80, EE + 2*HH, 1);   // W_ih.T rows 0..2559
    pack(W_hh,  pWl,   4*HH, 112, 80, 32, HH,   1);        // W_hh.T rows 2560..3583
    pack(W_pre, pWpre, HH,   112, 0,  112, 0,   2);        // row-permuted to [emb|ctx|h]
    pack(W_gen, pWgen, VV,   32,  0,  32, 0,    0);

    {   // bf16 encoder_hidden, combined bias, bridge init
        int n = BB * SS * 2 * HH;
        k_f32_to_bf16<<<dim3(n / 256), blk, 0, stream>>>(enc, ehB, n);
        k_addvec<<<dim3(16), blk, 0, stream>>>(b_ih, b_hh, bsum, 4 * HH);
        k_init_state<<<dim3(BB * HH / 256), blk, 0, stream>>>(ef_h, Wbh, bbh, hbuf,
                                                              xh + EE + 2 * HH);
        k_init_state<<<dim3(BB * HH / 256), blk, 0, stream>>>(ef_c, Wbc, bbc, cbuf,
                                                              (bfraw*)nullptr);
    }
    // proj_key = encoder_hidden @ Wk : [2048,2048]x[2048,1024]
    k_gemm<0><<<dim3(HH / 128, BB * SS / 32), blk, 0, stream>>>(
        ehB, 2 * HH, pWk, 64, HH, pkey, HH, nullptr);

    // ---- timestep loop ----
    for (int t = 0; t < TT; ++t) {
        // prev_embed gather (emb slot of xh)
        k_embed<<<dim3(BB), blk, 0, stream>>>(trg, emb, xh, t);
        // q = h_prev @ Wq (A = h slot of xh, carried from previous step / init)
        k_gemm<0><<<dim3(HH / 128, 1), blk, 0, stream>>>(
            xh + EE + 2 * HH, XDIM, pWq, 32, HH, qbuf, HH, nullptr);
        // energies + mask
        k_scores<<<dim3(BB * SS * 32 / 256), blk, 0, stream>>>(qbuf, pkey, ve, src_len, sbuf);
        // softmax over S + context into ctx slot of xh
        k_softmax_ctx<<<dim3(BB), blk, 0, stream>>>(sbuf, enc, xh);
        // LSTM gates: [32,3584] @ packed [3584,4096] + (b_ih+b_hh)
        k_gemm<1><<<dim3(4 * HH / 128, 1), blk, 0, stream>>>(
            xh, XDIM, pWl, 112, 4 * HH, gbuf, 4 * HH, bsum);
        // pointwise LSTM cell; writes h_new into h slot of xh
        k_lstm<<<dim3(BB * HH / 256), blk, 0, stream>>>(gbuf, hbuf, cbuf, xh);
        // pre = [emb|ctx|h_new] @ packed W_pre -> bf16
        k_gemm<2><<<dim3(HH / 128, 1), blk, 0, stream>>>(
            xh, XDIM, pWpre, 112, HH, preA, HH, nullptr);
        // logits = pre @ W_gen + b_gen  (dominant GEMM; weights L2-resident)
        k_gemm<1><<<dim3(VV / 128, 1), blk, 0, stream>>>(
            preA, HH, pWgen, 32, VV, lbuf, VV, b_gen);
        // vocab softmax -> out[b, v, t]
        k_softmax_v<<<dim3(BB), blk, 0, stream>>>(lbuf, out, t);
    }

    // ---- final hidden/cell state outputs ----
    k_copy_hc<<<dim3(BB * HH / 256), blk, 0, stream>>>(hbuf, cbuf, out + (size_t)BB * VV * TT);
}
